// BNL_19224273617285
// MI455X (gfx1250) — matrix operand order repaired and checked by reading, not compiled, not run on hardware
//
#include <hip/hip_runtime.h>
#include <hip/hip_bf16.h>
#include <cmath>

// ---------------------------------------------------------------------------
// CDNA5 (gfx1250, wave32) WMMA types: 16x16x32 bf16 -> f32
// ---------------------------------------------------------------------------
typedef __attribute__((ext_vector_type(16))) __bf16 bf16x16;
typedef __attribute__((ext_vector_type(8)))  float  floatx8;

// f32 -> bf16 bits (round-half-up; 1 add + 1 shift)
__device__ inline unsigned bfbits(float f) {
    union { float f; unsigned u; } v; v.f = f;
    return (v.u + 0x8000u) >> 16;
}
// bf16 bits (low 16 of u) -> f32
__device__ inline float bf2f(unsigned hu) {
    union { unsigned u; float f; } v; v.u = hu << 16; return v.f;
}
// pack two f32 into packed-bf16 dword: one v_perm_b32 after two adds
__device__ inline unsigned pack2bf(float lo, float hi) {
    union { float f; unsigned u; } a, b; a.f = lo; b.f = hi;
#if __has_builtin(__builtin_amdgcn_perm)
    // result bytes {3,2,1,0} = {hi[3],hi[2],lo[3],lo[2]}  (S1 = lo -> sel 2,3)
    return __builtin_amdgcn_perm(b.u + 0x8000u, a.u + 0x8000u, 0x07060302u);
#else
    return ((a.u + 0x8000u) >> 16) | ((b.u + 0x8000u) & 0xFFFF0000u);
#endif
}

// ---------------------------------------------------------------------------
// Fragment gathers.
// A (16x32 bf16) from LDS tile [M][sA] in contiguous k order (B staged to match):
//   lane l: row m = l&15; element j -> K = 16*(l>=16) + j
//   -> one contiguous 32B run per lane.
// ---------------------------------------------------------------------------
__device__ inline bf16x16 load_frag_a(const __bf16* ldsA, int sA, int m_base, int lane) {
    bf16x16 a;
    const int m    = m_base + (lane & 15);
    const int kofs = (lane >> 4) << 4;
#pragma unroll
    for (int j = 0; j < 16; ++j) {
        const int k = j + kofs;
        a[j] = ldsA[m * sA + k];
    }
    return a;
}

// B fragments staged fragment-major: element (k,n) of a 32x(16*T) tile lives at
// [tile(n>>4)][lane((n&15)|((k>>4)<<4))][j(k&15)]; each lane reads 32B
// contiguously -> 2x ds_load_b128.
__device__ inline bf16x16 load_frag_b_fm(const __bf16* ldsBf, int tile, int lane) {
    return *(const bf16x16*)(ldsBf + ((tile << 5) + lane) * 16);
}

// stage one packed row-pair (rows 2rp, 2rp+1; col c) into fragment-major LDS
__device__ inline void stage_fm(__bf16* ldsBf, int rp, int c, unsigned packed) {
    const int tile   = c >> 4;
    const int lane16 = (c & 15) | ((rp >> 3) << 4);
    const int j      = (rp & 7) << 1;
    *(unsigned*)(ldsBf + (tile << 9) + (lane16 << 4) + j) = packed;
}

// ---------------------------------------------------------------------------
// Kernel 1/2: batched GEMM  Y[b](bf16) = W(MxK,f32) @ X[b](KxN,f32)
// bf16 WMMA math, f32 accumulate. 256 threads = 8 waves (2M x 4N of 16x64).
// Double-buffered LDS, one barrier per k-step.
// ---------------------------------------------------------------------------
#define GT_MT 32
#define GT_NT 256
#define GT_KT 32

__global__ __launch_bounds__(256) void gemm_bf16_wmma(
    const float* __restrict__ W, const float* __restrict__ X,
    unsigned short* __restrict__ Y, int M, int N, int K,
    long long bsx, long long bsy)
{
    __shared__ __attribute__((aligned(32))) __bf16 ldsA[2][GT_MT][GT_KT + 8];
    __shared__ __attribute__((aligned(32))) __bf16 ldsBf[2][(GT_NT / 16) * 32 * 16];

    const int b  = blockIdx.z;
    const int n0 = blockIdx.x * GT_NT;
    const int m0 = blockIdx.y * GT_MT;
    const float* Xb = X + (long long)b * bsx;
    unsigned short* Yb = Y + (long long)b * bsy;

    const int tid  = threadIdx.x;
    const int lane = tid & 31;
    const int wave = tid >> 5;
    const int wm   = (wave >> 2) << 4;   // 0 or 16
    const int wn   = (wave & 3) << 6;    // 0,64,128,192

    auto stage = [&](int k0, int buf) {
        // A tile (32x32): thread owns (r = tid>>4 [+16], c = 2*(tid&15))
        {
            const int c  = (tid & 15) << 1;
            const int r0 = tid >> 4;
            const float* wa = &W[(long long)(m0 + r0) * K + k0 + c];
            const float2 w0 = *(const float2*)wa;
            const float2 w1 = *(const float2*)(wa + (long long)16 * K);
            *(unsigned*)&ldsA[buf][r0][c]      = pack2bf(w0.x, w0.y);
            *(unsigned*)&ldsA[buf][r0 + 16][c] = pack2bf(w1.x, w1.y);
        }
        // B tile (32x256): thread owns column c = tid, walks 16 row-pairs
        {
            const int c = tid;
            const float* xb = &Xb[(long long)k0 * N + n0 + c];
            const long long rowstep = 2LL * N;
            if (k0 + 2 * GT_KT < K) {          // uniform branch
                const long long pfoff = (long long)(2 * GT_KT) * N;
#pragma unroll
                for (int rp = 0; rp < GT_KT / 2; ++rp) {
                    const float v0 = xb[0];
                    const float v1 = xb[N];
                    __builtin_prefetch(xb + pfoff, 0, 1);  // global_prefetch_b8
                    stage_fm(&ldsBf[buf][0], rp, c, pack2bf(v0, v1));
                    xb += rowstep;
                }
            } else {
#pragma unroll
                for (int rp = 0; rp < GT_KT / 2; ++rp) {
                    const float v0 = xb[0];
                    const float v1 = xb[N];
                    stage_fm(&ldsBf[buf][0], rp, c, pack2bf(v0, v1));
                    xb += rowstep;
                }
            }
        }
    };

    floatx8 acc[4] = {};
    const int steps = K / GT_KT;
    stage(0, 0);
    for (int s = 0; s < steps; ++s) {
        __syncthreads();
        if (s + 1 < steps) stage((s + 1) * GT_KT, (s + 1) & 1);
        const int buf = s & 1;
        const bf16x16 afr = load_frag_a(&ldsA[buf][0][0], GT_KT + 8, wm, lane);
        bf16x16 bfr[4];
#pragma unroll
        for (int nf = 0; nf < 4; ++nf)
            bfr[nf] = load_frag_b_fm(&ldsBf[buf][0], (wn >> 4) + nf, lane);
#pragma unroll
        for (int nf = 0; nf < 4; ++nf)
            acc[nf] = __builtin_amdgcn_wmma_f32_16x16x32_bf16(
                false, afr, false, bfr[nf], (short)0, acc[nf], false, false);
    }

    // C layout: vgpr r -> row m = r + 8*(lane>=16), col n = lane&15
    const int mrow = (lane >= 16) ? 8 : 0;
    const int ncol = lane & 15;
#pragma unroll
    for (int nf = 0; nf < 4; ++nf) {
#pragma unroll
        for (int r = 0; r < 8; ++r) {
            Yb[(long long)(m0 + wm + mrow + r) * N + n0 + wn + (nf << 4) + ncol] =
                (unsigned short)bfbits(acc[nf][r]);
        }
    }
}

// ---------------------------------------------------------------------------
// Kernel 3: att[b,g,k] = alpha_k * sum_n p^k * g  over 32ch x 1024 spatial.
// ---------------------------------------------------------------------------
__global__ __launch_bounds__(256) void att_kernel(
    const unsigned short* __restrict__ p, const unsigned short* __restrict__ g,
    float* __restrict__ att, float a0, float a1, float a2)
{
    const int bg  = blockIdx.x;          // 0..127
    const int b   = bg >> 3, grp = bg & 7;
    const long long base = ((long long)b * 256 + grp * 32) * 1024;
    const unsigned* pp = (const unsigned*)(p + base);
    const unsigned* gp = (const unsigned*)(g + base);

    float s0 = 0.f, s1 = 0.f, s2 = 0.f;
    for (int i = threadIdx.x; i < 16 * 1024; i += 256) {
        const unsigned pu = pp[i], gu = gp[i];
        const float p0 = bf2f(pu & 0xFFFFu), p1 = bf2f(pu >> 16);
        const float g0 = bf2f(gu & 0xFFFFu), g1 = bf2f(gu >> 16);
        s0 += g0 + g1;
        s1 += p0 * g0 + p1 * g1;
        s2 += p0 * p0 * g0 + p1 * p1 * g1;
    }
    __shared__ float red0[256], red1[256], red2[256];
    red0[threadIdx.x] = s0; red1[threadIdx.x] = s1; red2[threadIdx.x] = s2;
    __syncthreads();
    for (int off = 128; off > 0; off >>= 1) {
        if (threadIdx.x < off) {
            red0[threadIdx.x] += red0[threadIdx.x + off];
            red1[threadIdx.x] += red1[threadIdx.x + off];
            red2[threadIdx.x] += red2[threadIdx.x + off];
        }
        __syncthreads();
    }
    if (threadIdx.x == 0) {
        att[bg * 3 + 0] = a0 * red0[0];
        att[bg * 3 + 1] = a1 * red1[0];
        att[bg * 3 + 2] = a2 * red2[0];
    }
}

// ---------------------------------------------------------------------------
// Kernel 4: per (b,g):  F(32xN) = c0 + c1*t + c2*t^2 (bf16, fragment-major),
//           Z(64xN) = Wz[g](64x32) @ F  -- K=32 == one WMMA per tile.
// Accumulates GroupNorm sum/sumsq (f32) into stats via atomics.
// 256 threads = 8 waves (4M x 2N tiles of 16x64), N tile 128.
// ---------------------------------------------------------------------------
#define ZT_NT 128

__global__ __launch_bounds__(256) void z_wmma_kernel(
    const unsigned short* __restrict__ t, const float* __restrict__ Wz,
    const float* __restrict__ att, unsigned short* __restrict__ z,
    float* __restrict__ stats, float a0, float a1, float a2)
{
    __shared__ __attribute__((aligned(32))) __bf16 ldsA[64][32 + 8];
    __shared__ __attribute__((aligned(32))) __bf16 ldsBf[(ZT_NT / 16) * 32 * 16];
    __shared__ float rsum[256], rsq[256];

    const int n0  = blockIdx.x * ZT_NT;
    const int grp = blockIdx.y;
    const int b   = blockIdx.z;
    const int bg  = b * 8 + grp;

    const float c0 = a0 * att[bg * 3 + 0];
    const float c1 = a1 * att[bg * 3 + 1];
    const float c2 = a2 * att[bg * 3 + 2];

    const int tid = threadIdx.x, lane = tid & 31, wave = tid >> 5;

    // stage Wz[g] (64x32): thread owns (c = 2*(tid&15)), rows tid>>4 step 16
    {
        const int c  = (tid & 15) << 1;
        const int r0 = tid >> 4;
        const float* wz = &Wz[(grp * 64 + r0) * 32 + c];
#pragma unroll
        for (int rr = 0; rr < 4; ++rr) {
            const float2 w2 = *(const float2*)(wz + rr * 16 * 32);
            *(unsigned*)&ldsA[r0 + rr * 16][c] = pack2bf(w2.x, w2.y);
        }
    }
    // stage t tile (32 x 128): thread owns col-pair c = 2*(tid&63),
    // row-pairs rp = tid>>6 step 4; poly applied in f32
    {
        const int c   = (tid & 63) << 1;
        const int rp0 = tid >> 6;                 // 0..3
        const unsigned short* tb =
            t + ((long long)b * 256 + grp * 32 + (rp0 << 1)) * 4096 + n0 + c;
#pragma unroll
        for (int rp = rp0; rp < 16; rp += 4) {
            const unsigned u0 = *(const unsigned*)tb;
            const unsigned u1 = *(const unsigned*)(tb + 4096);
            const float t00 = bf2f(u0 & 0xFFFFu), t01 = bf2f(u0 >> 16);
            const float t10 = bf2f(u1 & 0xFFFFu), t11 = bf2f(u1 >> 16);
            const float f00 = c0 + t00 * (c1 + t00 * c2);
            const float f01 = c0 + t01 * (c1 + t01 * c2);
            const float f10 = c0 + t10 * (c1 + t10 * c2);
            const float f11 = c0 + t11 * (c1 + t11 * c2);
            stage_fm(&ldsBf[0], rp, c,     pack2bf(f00, f10));
            stage_fm(&ldsBf[0], rp, c + 1, pack2bf(f01, f11));
            tb += 8LL * 4096;
        }
    }
    __syncthreads();

    const int wm = (wave >> 1) << 4;   // 0,16,32,48
    const int wn = (wave & 1) << 6;    // 0,64

    const bf16x16 afr = load_frag_a(&ldsA[0][0], 32 + 8, wm, lane);
    bf16x16 bfr[4];
#pragma unroll
    for (int nf = 0; nf < 4; ++nf)
        bfr[nf] = load_frag_b_fm(&ldsBf[0], (wn >> 4) + nf, lane);

    const long long zbase = ((long long)b * 512 + grp * 64) * 4096 + n0;
    const int mrow = (lane >= 16) ? 8 : 0;
    const int ncol = lane & 15;

    float s = 0.f, ss = 0.f;
#pragma unroll
    for (int nf = 0; nf < 4; ++nf) {
        floatx8 acc = {};
        acc = __builtin_amdgcn_wmma_f32_16x16x32_bf16(
            false, afr, false, bfr[nf], (short)0, acc, false, false);
#pragma unroll
        for (int r = 0; r < 8; ++r) {
            const float v = acc[r];
            z[zbase + (long long)(wm + mrow + r) * 4096 + wn + (nf << 4) + ncol] =
                (unsigned short)bfbits(v);
            s  += v;
            ss += v * v;
        }
    }

    rsum[tid] = s; rsq[tid] = ss;
    __syncthreads();
    for (int off = 128; off > 0; off >>= 1) {
        if (tid < off) {
            rsum[tid] += rsum[tid + off];
            rsq[tid]  += rsq[tid + off];
        }
        __syncthreads();
    }
    if (tid == 0) {
        atomicAdd(&stats[bg * 2 + 0], rsum[0]);
        atomicAdd(&stats[bg * 2 + 1], rsq[0]);
    }
}

// ---------------------------------------------------------------------------
// Kernel 5: GroupNorm normalize + affine + residual. 2 elements / thread.
// ---------------------------------------------------------------------------
__global__ __launch_bounds__(256) void finalize_kernel(
    const unsigned short* __restrict__ z, const float* __restrict__ stats,
    const float* __restrict__ gn_w, const float* __restrict__ gn_b,
    const float* __restrict__ xh, float* __restrict__ out)
{
    const long long e = ((long long)blockIdx.x * 256 + threadIdx.x) * 2;
    const long long per_b = 512LL * 4096LL;
    const int b   = (int)(e / per_b);
    const long long rem = e % per_b;
    const int ch  = (int)(rem >> 12);       // /4096
    const int grp = ch >> 6;                // /64
    const int bg  = b * 8 + grp;

    const float n   = 64.0f * 4096.0f;
    const float mu  = stats[bg * 2 + 0] / n;
    const float var = stats[bg * 2 + 1] / n - mu * mu;
    const float inv = rsqrtf(var + 1e-5f);
    const float w = gn_w[ch], bb = gn_b[ch];

    const unsigned zz = *(const unsigned*)(z + e);
    const float2 x2 = *(const float2*)(xh + e);
    float2 o;
    o.x = (bf2f(zz & 0xFFFFu) - mu) * inv * w + bb + x2.x;
    o.y = (bf2f(zz >> 16)     - mu) * inv * w + bb + x2.y;
    *(float2*)(out + e) = o;
}

// ---------------------------------------------------------------------------
// Host launcher
// ---------------------------------------------------------------------------
extern "C" void kernel_launch(void* const* d_in, const int* in_sizes, int n_in,
                              void* d_out, int out_size, void* d_ws, size_t ws_size,
                              hipStream_t stream) {
    const float* xh  = (const float*)d_in[0];   // [16,512,64,64]
    const float* xl  = (const float*)d_in[1];   // [16,512,32,32]
    const float* Wt  = (const float*)d_in[2];   // [256,512]
    const float* Wp  = (const float*)d_in[3];   // [256,512]
    const float* Wg  = (const float*)d_in[4];   // [256,512]
    const float* Wz  = (const float*)d_in[5];   // [8,64,32]
    const float* gnw = (const float*)d_in[6];   // [512]
    const float* gnb = (const float*)d_in[7];   // [512]
    float* out = (float*)d_out;

    // workspace carve-up (bf16 intermediates, f32 scalars)
    char* ws = (char*)d_ws;
    unsigned short* t_buf = (unsigned short*)ws;                       // 16*256*4096 bf16
    unsigned short* p_buf = (unsigned short*)(ws + 33554432);          // 16*256*1024 bf16
    unsigned short* g_buf = (unsigned short*)(ws + 33554432 + 8388608);
    float*          att   = (float*)(ws + 33554432 + 2 * 8388608);     // 384 f32
    unsigned short* z_buf = (unsigned short*)(ws + 33554432 + 2 * 8388608 + 4096);
    float*          stats = (float*)(ws + 33554432 + 2 * 8388608 + 4096 + 67108864);

    // Taylor alphas: alpha_k = sqrt((2g)^k / k! * exp(-2g)), g = 1e-4
    const double tg = 2.0e-4;
    const float a0 = (float)std::sqrt(std::exp(-tg));
    const float a1 = (float)std::sqrt(tg * std::exp(-tg));
    const float a2 = (float)std::sqrt(tg * tg * 0.5 * std::exp(-tg));

    hipMemsetAsync(stats, 0, 256 * sizeof(float), stream);

    // 1) t = Wt @ X_high   (M=256, K=512, N=4096, 16 batches)
    gemm_bf16_wmma<<<dim3(4096 / GT_NT, 256 / GT_MT, 16), 256, 0, stream>>>(
        Wt, xh, t_buf, 256, 4096, 512, 512LL * 4096, 256LL * 4096);
    // 2) p = Wp @ X_low, g = Wg @ X_low  (N=1024)
    gemm_bf16_wmma<<<dim3(1024 / GT_NT, 256 / GT_MT, 16), 256, 0, stream>>>(
        Wp, xl, p_buf, 256, 1024, 512, 512LL * 1024, 256LL * 1024);
    gemm_bf16_wmma<<<dim3(1024 / GT_NT, 256 / GT_MT, 16), 256, 0, stream>>>(
        Wg, xl, g_buf, 256, 1024, 512, 512LL * 1024, 256LL * 1024);
    // 3) attention coefficients
    att_kernel<<<dim3(128), 256, 0, stream>>>(p_buf, g_buf, att, a0, a1, a2);
    // 4) z = Wz @ poly(t) + GroupNorm stats
    z_wmma_kernel<<<dim3(4096 / ZT_NT, 8, 16), 256, 0, stream>>>(
        t_buf, Wz, att, z_buf, stats, a0, a1, a2);
    // 5) GroupNorm + residual (2 elems/thread)
    finalize_kernel<<<dim3((unsigned)(16LL * 512 * 4096 / 512)), 256, 0, stream>>>(
        z_buf, stats, gnw, gnb, xh, out);

    (void)in_sizes; (void)n_in; (void)out_size; (void)ws_size;
}
